// SGM_54889682043709
// MI455X (gfx1250) — matrix operand order repaired and checked by reading, not compiled, not run on hardware
//
#include <hip/hip_runtime.h>
#include <cstdint>

// ---------------------------------------------------------------------------
// MI455X (gfx1250) SMPL graph-conv head.
//
// ~63 GFLOP vs ~83 MB HBM -> ~760 FLOP/B, compute bound.  Strategy:
//   * fold shape branch (w_shape @ W_mlp1) once -> removes 40% of FLOPs
//   * pre-format all weights ONCE into f16 B-operand-layout images in d_ws;
//     main kernel pulls them with GLOBAL_LOAD_ASYNC_TO_LDS_B128 (ASYNCcnt DMA)
//     overlapped with x staging
//   * all GEMMs on v_wmma_f32_16x16x32_f16 (f32 accumulate), everything in LDS
// ---------------------------------------------------------------------------

typedef _Float16 half_t;
typedef __attribute__((ext_vector_type(8)))  _Float16 h8;
typedef __attribute__((ext_vector_type(16))) _Float16 v16h;
typedef __attribute__((ext_vector_type(8)))  float    v8f;
typedef __attribute__((ext_vector_type(4)))  int      v4i;

#define BT   4      // batch elements per workgroup
#define ROWS 96     // BT * 24 node-rows
#define PX   72     // LDS pitch (halfs) of x buffer (K padded 48->64)
#define PH   136    // LDS pitch (halfs) of 128-feature buffers
#define PW   136    // pitch (halfs) of weight rows, B-operand layout [n][k]

// Pre-formatted weight image offsets (halfs) inside d_ws
#define IMG_E 0                    // embed  [128][PW]
#define IMG_1 17408                // pose1  [256][PW]  (c*128+n rows)
#define IMG_F 52224                // final  [64][PW]   (w2 rows 0..31, Wf rows 32..63)
#define IMG_TOT 60928              // total halfs; bf[10] f32 follows

#if __has_builtin(__builtin_amdgcn_global_load_async_to_lds_b128) && \
    __has_builtin(__builtin_amdgcn_s_wait_asynccnt)
#define USE_ASYNC 1
#else
#define USE_ASYNC 0
#endif

// Padded 4-neighbor adjacency of the 24-joint skeleton, row-normalized
// (weight = 1/sqrt(deg), zero-weight padding slots).
static __device__ const int NBR[24][4] = {
  {1,2,3,0},  {0,4,0,0},  {0,5,0,0},  {0,6,0,0},
  {1,7,0,0},  {2,8,0,0},  {3,9,0,0},  {4,10,0,0},
  {5,11,0,0}, {6,12,13,14},{7,0,0,0}, {8,0,0,0},
  {9,15,0,0}, {9,16,0,0}, {9,17,0,0}, {12,0,0,0},
  {13,18,0,0},{14,19,0,0},{16,20,0,0},{17,21,0,0},
  {18,22,0,0},{19,23,0,0},{20,0,0,0}, {21,0,0,0}
};
static __device__ const float NWS[24][4] = {
  {0.57735027f,0.57735027f,0.57735027f,0.f},
  {0.70710678f,0.70710678f,0.f,0.f},{0.70710678f,0.70710678f,0.f,0.f},
  {0.70710678f,0.70710678f,0.f,0.f},{0.70710678f,0.70710678f,0.f,0.f},
  {0.70710678f,0.70710678f,0.f,0.f},{0.70710678f,0.70710678f,0.f,0.f},
  {0.70710678f,0.70710678f,0.f,0.f},{0.70710678f,0.70710678f,0.f,0.f},
  {0.5f,0.5f,0.5f,0.5f},
  {1.f,0.f,0.f,0.f},{1.f,0.f,0.f,0.f},
  {0.70710678f,0.70710678f,0.f,0.f},{0.70710678f,0.70710678f,0.f,0.f},
  {0.70710678f,0.70710678f,0.f,0.f},
  {1.f,0.f,0.f,0.f},
  {0.70710678f,0.70710678f,0.f,0.f},{0.70710678f,0.70710678f,0.f,0.f},
  {0.70710678f,0.70710678f,0.f,0.f},{0.70710678f,0.70710678f,0.f,0.f},
  {0.70710678f,0.70710678f,0.f,0.f},{0.70710678f,0.70710678f,0.f,0.f},
  {1.f,0.f,0.f,0.f},{1.f,0.f,0.f,0.f}
};

// Copy nbytes (multiple of 16, 16B-aligned both sides) global -> LDS.
static __device__ __forceinline__ void copy16(int tid, half_t* lds,
                                              const half_t* g, int nbytes) {
#if USE_ASYNC
  // GLOBAL_LOAD_ASYNC_TO_LDS_B128: 16B per lane, tracked by ASYNCcnt.
  // Generic->AS3: hardware maps generic LDS addresses via addr[31:0].
  for (int i = tid * 16; i < nbytes; i += 256 * 16) {
    unsigned lofs = (unsigned)(uintptr_t)lds + (unsigned)i;
    __builtin_amdgcn_global_load_async_to_lds_b128(
        (__attribute__((address_space(1))) v4i*)((uintptr_t)g + i),
        (__attribute__((address_space(3))) v4i*)(uintptr_t)lofs, 0, 0);
  }
#else
  const uint4* gp = (const uint4*)g;
  uint4*       lp = (uint4*)lds;
  for (int i = tid; i < nbytes / 16; i += 256) lp[i] = gp[i];
#endif
}

// One 16x16 tile over K (multiple of 32): load all fragments, then WMMA chain.
// A halfs i -> K = (i/8)*16 + hi*8 + i%8 (two b128); B halfs i -> same K.
template<int K>
static __device__ __forceinline__ v8f gemm_acc(const half_t* __restrict__ A, int aOff,
                                               const half_t* __restrict__ B, int bOff,
                                               int hi, v8f acc) {
  constexpr int NT = K / 32;
  union U { v16h v; h8 p[2]; };
  U a[NT], b[NT];
#pragma unroll
  for (int t = 0; t < NT; ++t) {
    const int k0 = t * 32;
    a[t].p[0] = *(const h8*)(A + aOff + k0 + hi * 8);
    a[t].p[1] = *(const h8*)(A + aOff + k0 + hi * 8 + 16);
    b[t].p[0] = *(const h8*)(B + bOff + k0 + hi * 8);
    b[t].p[1] = *(const h8*)(B + bOff + k0 + hi * 8 + 16);
  }
#pragma unroll
  for (int t = 0; t < NT; ++t)
    acc = __builtin_amdgcn_wmma_f32_16x16x32_f16(false, a[t].v, false, b[t].v,
                                                 (short)0, acc, false, false);
  return acc;
}

// ---------------------------------------------------------------------------
// One-time weight prep (f32-exact): builds f16 B-layout images + folded shape
// branch  Wf[c][f][q] = sum_o w_shape[c][f][o]*Wm1[o][q],
//         bf[q] = sum_o b_shape[o]*Wm1[o][q] + bm1[q].
// ---------------------------------------------------------------------------
__global__ __launch_bounds__(256) void prep_weights(
    const float* __restrict__ We, const float* __restrict__ w1,
    const float* __restrict__ w2, const float* __restrict__ ws,
    const float* __restrict__ bs, const float* __restrict__ Wm1,
    const float* __restrict__ bm1, half_t* __restrict__ img,
    float* __restrict__ bf) {
  int t = blockIdx.x * 256 + threadIdx.x;
  if (t < IMG_TOT) {
    float v = 0.0f;
    if (t < IMG_1) {                               // embed [128][PW]
      int n = t / PW, k = t % PW;
      if (k < 48) v = We[k * 128 + n];
    } else if (t < IMG_F) {                        // pose1 [256][PW]
      int u = t - IMG_1, r = u / PW, k = u % PW;
      int c = r >> 7, n = r & 127;
      if (k < 128) v = w1[((size_t)(c * 128 + k)) * 128 + n];
    } else {                                       // final [64][PW]
      int u = t - IMG_F, r = u / PW, k = u % PW;
      if (k < 128) {
        if (r < 32) {                              // w_pose2, N padded 4->16
          int c = r >> 4, n = r & 15;
          if (n < 4) v = w2[(c * 128 + k) * 4 + n];
        } else {                                   // fused Wf, N padded 10->16
          int rr = r - 32, c = rr >> 4, q = rr & 15;
          if (q < 10) {
            float s = 0.0f;
            for (int o = 0; o < 128; ++o)
              s += ws[((size_t)c * 128 + k) * 128 + o] * Wm1[o * 10 + q];
            v = s;
          }
        }
      }
    }
    img[t] = (half_t)v;
  }
  if (t < 10) {
    float s = bm1[t];
    for (int o = 0; o < 128; ++o) s += bs[o] * Wm1[o * 10 + t];
    bf[t] = s;
  }
}

__global__ __launch_bounds__(256) void smpl_head_kernel(
    const float* __restrict__ x,
    const float* __restrict__ be, const float* __restrict__ b1,
    const float* __restrict__ b2,
    const half_t* __restrict__ img, const float* __restrict__ bf,
    const float* __restrict__ Wm2, const float* __restrict__ bm2,
    float* __restrict__ pose, float* __restrict__ shapeOut, int Bn)
{
  extern __shared__ char smem[];
  half_t* sX   = (half_t*)smem;            // ROWS*PX  (x f16; dead after embed)
  half_t* sH   = sX   + ROWS * PX;         // ROWS*PH  h
  half_t* sX1  = sH   + ROWS * PH;         // ROWS*PH  gso@h
  half_t* sPF  = sX1  + ROWS * PH;         // ROWS*PH  pose feature
  half_t* sXPF = sPF  + ROWS * PH;         // ROWS*PH  gso@pf
  half_t* sWE  = sXPF + ROWS * PH;         // 128*PW   embed weights
  half_t* sW1  = sWE  + 128 * PW;          // 256*PW   pose1 weights
  half_t* sWF  = sW1  + 256 * PW;          // 64*PW    pose2 + fused shape
  float*  sSF  = (float*)sX;               // ROWS*12 f32 (aliases dead sX)

  const int tid  = threadIdx.x;
  const int wave = tid >> 5;
  const int lane = tid & 31;
  const int ln   = lane & 15;              // N index within tile
  const int hi   = lane >> 4;              // lane-half select
  const int b0   = blockIdx.x * BT;

  // ---- Phase 0: kick off weight DMA (async), stage x meanwhile -----------
  copy16(tid, sWE, img + IMG_E, 128 * PW * 2);
  copy16(tid, sW1, img + IMG_1, 256 * PW * 2);
  copy16(tid, sWF, img + IMG_F, 64 * PW * 2);

  for (int i = tid; i < ROWS * 64; i += 256) {
    int r = i >> 6, f = i & 63;
    int bb = b0 + r / 24; if (bb >= Bn) bb = Bn - 1;
    float v = (f < 48) ? x[((size_t)bb * 24 + (r % 24)) * 48 + f] : 0.0f;
    sX[r * PX + f] = (half_t)v;
  }
#if USE_ASYNC
  __builtin_amdgcn_s_wait_asynccnt(0);
#endif
  __syncthreads();

  // ---- Phase 1: h = x @ We + be  (6 row tiles x 8 col tiles) -------------
  {
    const int n = wave * 16 + ln;
    const float bv = be[n];
    for (int rt = 0; rt < 6; ++rt) {
      v8f acc;
#pragma unroll
      for (int i = 0; i < 8; ++i) acc[i] = bv;
      acc = gemm_acc<64>(sX, (rt * 16 + ln) * PX, sWE, n * PW, hi, acc);
#pragma unroll
      for (int i = 0; i < 8; ++i)
        sH[(rt * 16 + hi * 8 + i) * PH + n] = (half_t)acc[i];
    }
  }
  __syncthreads();

  // ---- Phase 2: x1 = gso (.) h ------------------------------------------
  for (int i = tid; i < ROWS * 128; i += 256) {
    int r = i >> 7, f = i & 127;
    int bb = r / 24, nn = r % 24;
    float s = 0.0f;
#pragma unroll
    for (int sl = 0; sl < 4; ++sl)
      s += NWS[nn][sl] * (float)sH[(bb * 24 + NBR[nn][sl]) * PH + f];
    sX1[r * PH + f] = (half_t)s;
  }
  __syncthreads();

  // ---- Phase 3: pf = h @ w1[0] + x1 @ w1[1] + b1  (K=256 concat) ---------
  {
    const int n = wave * 16 + ln;
    const float bv = b1[n];
    for (int rt = 0; rt < 6; ++rt) {
      v8f acc;
#pragma unroll
      for (int i = 0; i < 8; ++i) acc[i] = bv;
      const int aOff = (rt * 16 + ln) * PH;
      acc = gemm_acc<128>(sH,  aOff, sW1, n * PW,         hi, acc);
      acc = gemm_acc<128>(sX1, aOff, sW1, (128 + n) * PW, hi, acc);
#pragma unroll
      for (int i = 0; i < 8; ++i)
        sPF[(rt * 16 + hi * 8 + i) * PH + n] = (half_t)acc[i];
    }
  }
  __syncthreads();

  // ---- Phase 4: x1pf = gso (.) pf ----------------------------------------
  for (int i = tid; i < ROWS * 128; i += 256) {
    int r = i >> 7, f = i & 127;
    int bb = r / 24, nn = r % 24;
    float s = 0.0f;
#pragma unroll
    for (int sl = 0; sl < 4; ++sl)
      s += NWS[nn][sl] * (float)sPF[(bb * 24 + NBR[nn][sl]) * PH + f];
    sXPF[r * PH + f] = (half_t)s;
  }
  __syncthreads();

  // ---- Phase 5: 12 tile jobs over 8 waves --------------------------------
  //   jobs 0..5 : pose = pf @ w2[0] + x1pf @ w2[1] + b2  -> global
  //   jobs 6..11: sf10 = h @ Wf[0] + x1 @ Wf[1] + bf     -> sSF
  for (int j = wave; j < 12; j += 8) {
    const int rt = (j < 6) ? j : (j - 6);
    const int aOff = (rt * 16 + ln) * PH;
    v8f acc;
    if (j < 6) {
      const float bv = (ln < 4) ? b2[ln] : 0.0f;
#pragma unroll
      for (int i = 0; i < 8; ++i) acc[i] = bv;
      acc = gemm_acc<128>(sPF,  aOff, sWF, ln * PW,        hi, acc);
      acc = gemm_acc<128>(sXPF, aOff, sWF, (16 + ln) * PW, hi, acc);
      if (ln < 4) {
#pragma unroll
        for (int i = 0; i < 8; ++i) {
          int m = rt * 16 + hi * 8 + i;
          int bb = b0 + m / 24;
          if (bb < Bn)
            pose[(((size_t)bb) * 24 + (m % 24)) * 4 + ln] = acc[i];
        }
      }
    } else {
      const float bv = (ln < 10) ? bf[ln] : 0.0f;
#pragma unroll
      for (int i = 0; i < 8; ++i) acc[i] = bv;
      acc = gemm_acc<128>(sH,  aOff, sWF, (32 + ln) * PW, hi, acc);
      acc = gemm_acc<128>(sX1, aOff, sWF, (48 + ln) * PW, hi, acc);
      if (ln < 10) {
#pragma unroll
        for (int i = 0; i < 8; ++i)
          sSF[(rt * 16 + hi * 8 + i) * 12 + ln] = acc[i];
      }
    }
  }
  __syncthreads();

  // ---- Phase 6: shape = reshape(sf10) @ Wm2 + bm2 ------------------------
  if (tid < 40) {
    int bb = tid / 10, q = tid % 10;
    if (b0 + bb < Bn) {
      float s = bm2[q];
      for (int j = 0; j < 240; ++j)
        s += sSF[(bb * 24 + j / 10) * 12 + (j % 10)] * Wm2[j * 10 + q];
      shapeOut[((size_t)(b0 + bb)) * 10 + q] = s;
    }
  }
}

extern "C" void kernel_launch(void* const* d_in, const int* in_sizes, int n_in,
                              void* d_out, int out_size, void* d_ws, size_t ws_size,
                              hipStream_t stream) {
  (void)n_in; (void)out_size; (void)ws_size;
  const float* x   = (const float*)d_in[0];
  const float* We  = (const float*)d_in[1];
  const float* be  = (const float*)d_in[2];
  const float* w1  = (const float*)d_in[3];
  const float* b1  = (const float*)d_in[4];
  const float* w2  = (const float*)d_in[5];
  const float* b2  = (const float*)d_in[6];
  const float* ws  = (const float*)d_in[7];
  const float* bs  = (const float*)d_in[8];
  const float* Wm1 = (const float*)d_in[9];
  const float* bm1 = (const float*)d_in[10];
  const float* Wm2 = (const float*)d_in[11];
  const float* bm2 = (const float*)d_in[12];

  const int Bn = in_sizes[0] / (24 * 48);
  float*  pose  = (float*)d_out;
  float*  shape = pose + (size_t)Bn * 24 * 4;
  half_t* img   = (half_t*)d_ws;                       // 60928 halfs
  float*  bf    = (float*)((char*)d_ws + IMG_TOT * 2); // 10 floats

  prep_weights<<<(IMG_TOT + 255) / 256, 256, 0, stream>>>(We, w1, w2, ws, bs,
                                                          Wm1, bm1, img, bf);

  const int nblk = (Bn + BT - 1) / BT;
  const size_t smem =
      (size_t)(ROWS * PX + 4 * ROWS * PH + (128 + 256 + 64) * PW) * sizeof(half_t);
  smpl_head_kernel<<<nblk, 256, smem, stream>>>(x, be, b1, b2, img, bf,
                                                Wm2, bm2, pose, shape, Bn);
}
